// SageConvLayer_154618823108
// MI455X (gfx1250) — matrix operand order repaired and checked by reading, not compiled, not run on hardware
//
#include <hip/hip_runtime.h>

// ---- types ---------------------------------------------------------------
typedef __attribute__((ext_vector_type(16))) __bf16      v16bf;
typedef __attribute__((ext_vector_type(8)))  float       v8f;
typedef __attribute__((ext_vector_type(4)))  float       f32x4;
typedef __attribute__((ext_vector_type(4)))  unsigned int u32x4;
typedef __attribute__((ext_vector_type(2)))  unsigned int u32x2;
typedef __attribute__((ext_vector_type(4)))  int          i32x4;

// AS-qualified int4 for the async-copy builtin (signature learned from hipcc):
//   param0: AS1 (global) i32x4*, param1: AS3 (LDS) i32x4*, then imm offset, imm cpol
typedef __attribute__((address_space(1))) i32x4 gbl_i32x4;
typedef __attribute__((address_space(3))) i32x4 lds_i32x4;

union Frag16 {
    v16bf bf;
    u32x4 q[2];
    unsigned short s[16];
};

#define NN   8192
#define FIN  512
#define FOUT 512
#define K2   1024   // 2*FIN

#if __has_builtin(__builtin_amdgcn_global_load_async_to_lds_b128)
#define ASYNC_OK 1
#else
#define ASYNC_OK 0
#endif

__device__ __forceinline__ void wait_asynccnt0() {
#if __has_builtin(__builtin_amdgcn_s_wait_asynccnt)
    __builtin_amdgcn_s_wait_asynccnt(0);
#else
    asm volatile("s_wait_asynccnt 0x0" ::: "memory");
#endif
}

__device__ __forceinline__ unsigned int f2bf(float f) {
    unsigned int u = __float_as_uint(f);
    return (u + 0x7FFFu + ((u >> 16) & 1u)) >> 16;   // RNE f32->bf16, low 16 bits valid
}

// ---- kernel: W (f32 [512][1024]) -> bf16 row-major -----------------------
__global__ void __launch_bounds__(256) k_cvt_w(const float* __restrict__ W,
                                               unsigned short* __restrict__ WB) {
    int gid = blockIdx.x * 256 + threadIdx.x;          // 131072 threads, 4 elems each
    f32x4 w = *(const f32x4*)(W + (size_t)gid * 4);
    u32x2 o;
    o.x = (f2bf(w.x) & 0xFFFFu) | (f2bf(w.y) << 16);
    o.y = (f2bf(w.z) & 0xFFFFu) | (f2bf(w.w) << 16);
    *(u32x2*)(WB + (size_t)gid * 4) = o;
}

// ---- kernel: features -> bf16 into data[:, 0:512] and featT (transposed) -
__global__ void __launch_bounds__(256) k_cvt_feat(const float* __restrict__ F,
                                                  unsigned short* __restrict__ dataB,
                                                  unsigned short* __restrict__ featT) {
    int gid = blockIdx.x * 256 + threadIdx.x;          // N*FIN/4 threads
    int n = gid >> 7;                                  // 512/4 = 128 quads per row
    int c = (gid & 127) << 2;
    f32x4 f = *(const f32x4*)(F + (size_t)n * FIN + c);
    unsigned int b0 = f2bf(f.x) & 0xFFFFu, b1 = f2bf(f.y) & 0xFFFFu;
    unsigned int b2 = f2bf(f.z) & 0xFFFFu, b3 = f2bf(f.w) & 0xFFFFu;
    u32x2 o; o.x = b0 | (b1 << 16); o.y = b2 | (b3 << 16);
    *(u32x2*)(dataB + (size_t)n * K2 + c) = o;
    featT[(size_t)(c + 0) * NN + n] = (unsigned short)b0;
    featT[(size_t)(c + 1) * NN + n] = (unsigned short)b1;
    featT[(size_t)(c + 2) * NN + n] = (unsigned short)b2;
    featT[(size_t)(c + 3) * NN + n] = (unsigned short)b3;
}

// ---- GEMM1: neigh = (adj @ features) / deg, fused row-degree -------------
// Block: 256 thr (8 waves). BM=32, BN=512 (64 cols/wave), BK=64, 256 blocks.
// Double-buffered LDS; next f32 adj tile prefetched to VGPRs during compute.
__global__ void __launch_bounds__(256) k_gemm1(const float* __restrict__ adj,
                                               const unsigned short* __restrict__ featT,
                                               unsigned short* __restrict__ dataB) {
    __shared__ alignas(16) unsigned short As[2][32][72];  // 2 x (32x64 bf16, padded)
    __shared__ float degs[32];

    const int tid  = threadIdx.x;
    const int lane = tid & 31;
    const int wv   = tid >> 5;                 // 0..7
    const int m0   = blockIdx.x * 32;
    const int sRow = tid >> 3;                 // staging: row 0..31
    const int sCol = (tid & 7) << 3;           // staging: col 0..56 (8 elems)

    if (tid < 32) degs[tid] = 1.0f;            // "+1" of the reference
    const float* arow = adj + (size_t)(m0 + sRow) * NN + sCol;

    v8f acc[2][4];
#pragma unroll
    for (int mt = 0; mt < 2; ++mt)
#pragma unroll
        for (int nt = 0; nt < 4; ++nt)
            acc[mt][nt] = (v8f){0.f,0.f,0.f,0.f,0.f,0.f,0.f,0.f};

    float degacc = 0.0f;
    const int nb = wv * 64;

    // prologue: first adj tile into registers
    f32x4 r0 = *(const f32x4*)(arow);
    f32x4 r1 = *(const f32x4*)(arow + 4);

    int buf = 0;
    for (int k0 = 0; k0 < NN; k0 += 64) {
        // convert staged registers -> bf16 LDS tile, fold into degree sum
        degacc += (r0.x + r0.y) + (r0.z + r0.w) + (r1.x + r1.y) + (r1.z + r1.w);
        u32x4 p;
        p.x = (f2bf(r0.x) & 0xFFFFu) | (f2bf(r0.y) << 16);
        p.y = (f2bf(r0.z) & 0xFFFFu) | (f2bf(r0.w) << 16);
        p.z = (f2bf(r1.x) & 0xFFFFu) | (f2bf(r1.y) << 16);
        p.w = (f2bf(r1.z) & 0xFFFFu) | (f2bf(r1.w) << 16);
        *(u32x4*)&As[buf][sRow][sCol] = p;
        __syncthreads();

        // prefetch next adj tile into registers (overlaps with WMMAs below)
        if (k0 + 64 < NN) {
            r0 = *(const f32x4*)(arow + k0 + 64);
            r1 = *(const f32x4*)(arow + k0 + 68);
        }

#pragma unroll
        for (int kk = 0; kk < 64; kk += 32) {
            Frag16 b[4];
#pragma unroll
            for (int nt = 0; nt < 4; ++nt) {
                // B col-major: featT[f][n], lane covers K 8h..8h+7 and 16+8h..16+8h+7 (=node idx)
                const unsigned short* bp =
                    featT + (size_t)(nb + nt * 16 + (lane & 15)) * NN
                          + (k0 + kk + ((lane >> 4) << 3));
                b[nt].q[0] = *(const u32x4*)bp;
                b[nt].q[1] = *(const u32x4*)(bp + 16);
            }
#pragma unroll
            for (int mt = 0; mt < 2; ++mt) {
                Frag16 a;
                const int row = mt * 16 + (lane & 15);
                const int kb  = kk + ((lane >> 4) << 3);
                a.q[0] = *(const u32x4*)&As[buf][row][kb];
                a.q[1] = *(const u32x4*)&As[buf][row][kb + 16];
#pragma unroll
                for (int nt = 0; nt < 4; ++nt)
                    acc[mt][nt] = __builtin_amdgcn_wmma_f32_16x16x32_bf16(
                        false, a.bf, false, b[nt].bf, (short)0, acc[mt][nt],
                        false, false);
            }
        }
        buf ^= 1;
    }

    atomicAdd(&degs[sRow], degacc);            // ds_add_f32
    __syncthreads();

    // scale by 1/deg, convert, store into data[:, 512:1024]
#pragma unroll
    for (int mt = 0; mt < 2; ++mt) {
#pragma unroll
        for (int v = 0; v < 8; ++v) {
            const int rl  = mt * 16 + ((lane >> 4) << 3) + v;  // C layout: M = v (+8 hi lanes)
            const float inv = 1.0f / degs[rl];
#pragma unroll
            for (int nt = 0; nt < 4; ++nt) {
                const int col = nb + nt * 16 + (lane & 15);
                dataB[(size_t)(m0 + rl) * K2 + FIN + col] =
                    (unsigned short)f2bf(acc[mt][nt][v] * inv);
            }
        }
    }
}

// ---- GEMM2: out = data @ W.T  (data bf16 [8192][1024], W bf16 [512][1024])
// A tile staged with async global->LDS copies (ASYNCcnt), double-buffered.
__global__ void __launch_bounds__(256) k_gemm2(const unsigned short* __restrict__ dataB,
                                               const unsigned short* __restrict__ WB,
                                               float* __restrict__ out) {
    __shared__ alignas(16) unsigned short As[2][32][72];

    const int tid  = threadIdx.x;
    const int lane = tid & 31;
    const int wv   = tid >> 5;
    const int m0   = blockIdx.x * 32;
    const int sRow = tid >> 3;
    const int sCol = (tid & 7) << 3;

    const unsigned short* arow = dataB + (size_t)(m0 + sRow) * K2 + sCol;

    v8f acc[2][4];
#pragma unroll
    for (int mt = 0; mt < 2; ++mt)
#pragma unroll
        for (int nt = 0; nt < 4; ++nt)
            acc[mt][nt] = (v8f){0.f,0.f,0.f,0.f,0.f,0.f,0.f,0.f};

    const int nb = wv * 64;

#if ASYNC_OK
    // prologue: async-stage tile 0 (16B per thread, 4KB tile)
    __builtin_amdgcn_global_load_async_to_lds_b128(
        (gbl_i32x4*)(arow),
        (lds_i32x4*)&As[0][sRow][sCol], 0, 0);
    wait_asynccnt0();
    __syncthreads();
#else
    u32x4 r = *(const u32x4*)(arow);
#endif

    int buf = 0;
    for (int k0 = 0; k0 < K2; k0 += 64) {
#if ASYNC_OK
        // kick off async copy of next tile into the other buffer
        if (k0 + 64 < K2)
            __builtin_amdgcn_global_load_async_to_lds_b128(
                (gbl_i32x4*)(arow + k0 + 64),
                (lds_i32x4*)&As[buf ^ 1][sRow][sCol], 0, 0);
#else
        *(u32x4*)&As[buf][sRow][sCol] = r;
        __syncthreads();
        if (k0 + 64 < K2) r = *(const u32x4*)(arow + k0 + 64);
#endif

#pragma unroll
        for (int kk = 0; kk < 64; kk += 32) {
            Frag16 b[4];
#pragma unroll
            for (int nt = 0; nt < 4; ++nt) {
                // B = W^T col-major == W row-major: WB[o][k], K 8h..8h+7 and 16+8h..16+8h+7 per lane
                const unsigned short* bp =
                    WB + (size_t)(nb + nt * 16 + (lane & 15)) * K2
                       + (k0 + kk + ((lane >> 4) << 3));
                b[nt].q[0] = *(const u32x4*)bp;
                b[nt].q[1] = *(const u32x4*)(bp + 16);
            }
#pragma unroll
            for (int mt = 0; mt < 2; ++mt) {
                Frag16 a;
                const int row = mt * 16 + (lane & 15);
                const int kb  = kk + ((lane >> 4) << 3);
                a.q[0] = *(const u32x4*)&As[buf][row][kb];
                a.q[1] = *(const u32x4*)&As[buf][row][kb + 16];
#pragma unroll
                for (int nt = 0; nt < 4; ++nt)
                    acc[mt][nt] = __builtin_amdgcn_wmma_f32_16x16x32_bf16(
                        false, a.bf, false, b[nt].bf, (short)0, acc[mt][nt],
                        false, false);
            }
        }

#if ASYNC_OK
        wait_asynccnt0();        // own async copy (next tile) complete
        __syncthreads();         // publish to all waves
#endif
        buf ^= 1;
    }

#pragma unroll
    for (int mt = 0; mt < 2; ++mt)
#pragma unroll
        for (int v = 0; v < 8; ++v) {
            const int rl = mt * 16 + ((lane >> 4) << 3) + v;
#pragma unroll
            for (int nt = 0; nt < 4; ++nt) {
                const int col = nb + nt * 16 + (lane & 15);
                out[(size_t)(m0 + rl) * FOUT + col] = acc[mt][nt][v];
            }
        }
}

// ---- launcher ------------------------------------------------------------
extern "C" void kernel_launch(void* const* d_in, const int* in_sizes, int n_in,
                              void* d_out, int out_size, void* d_ws, size_t ws_size,
                              hipStream_t stream) {
    (void)in_sizes; (void)n_in; (void)out_size; (void)ws_size;
    const float* adj  = (const float*)d_in[0];
    const float* feat = (const float*)d_in[1];
    const float* W    = (const float*)d_in[2];
    float* out = (float*)d_out;

    char* ws = (char*)d_ws;
    unsigned short* WB    = (unsigned short*)ws;                               // 512*1024 bf16 (1 MB)
    unsigned short* featT = (unsigned short*)(ws + (size_t)FOUT * K2 * 2);     // 512*8192 bf16 (8 MB)
    unsigned short* dataB = (unsigned short*)(ws + (size_t)FOUT * K2 * 2
                                                 + (size_t)FIN * NN * 2);      // 8192*1024 bf16 (16 MB)

    k_cvt_w   <<<(FOUT * K2 / 4 + 255) / 256, 256, 0, stream>>>(W, WB);
    k_cvt_feat<<<(NN * FIN / 4 + 255) / 256, 256, 0, stream>>>(feat, dataB, featT);
    k_gemm1   <<<NN / 32, 256, 0, stream>>>(adj, featT, dataB);
    k_gemm2   <<<NN / 32, 256, 0, stream>>>(dataB, WB, out);
}
